// SelfAttention_68401649156637
// MI455X (gfx1250) — compile-verified, matrix operand order checked by reading, not run on hardware
//
#include <hip/hip_runtime.h>

// ---------- types ----------
typedef __attribute__((ext_vector_type(16))) __bf16        v16bf;
typedef __attribute__((ext_vector_type(8)))  float         v8f;
typedef __attribute__((ext_vector_type(4)))  unsigned int  u32x4;
typedef __attribute__((ext_vector_type(4)))  unsigned int  v4u;
typedef __attribute__((ext_vector_type(8)))  int           v8i;
typedef __attribute__((ext_vector_type(4)))  int           v4i;

union BF16Frag { v16bf v; u32x4 u[2]; unsigned short s[16]; };

#define NTOK 2048
#define DIMV 1024
#define HEADS 16
#define DH 64
#define CH 64
#define NCH (NTOK / CH)

#if defined(__has_builtin)
#if __has_builtin(__builtin_amdgcn_tensor_load_to_lds) && __has_builtin(__builtin_amdgcn_s_wait_tensorcnt)
#define HAVE_TDM 1
#endif
#endif

// ---------- helpers ----------
static __device__ __forceinline__ unsigned short f2bf(float f) {
  unsigned int u = __float_as_uint(f);
  u += 0x7FFFu + ((u >> 16) & 1u);           // round-to-nearest-even
  return (unsigned short)(u >> 16);
}
static __device__ __forceinline__ float bf2f(unsigned short h) {
  return __uint_as_float(((unsigned int)h) << 16);
}

// Load one 16x32 bf16 WMMA operand fragment from a row-major [rows][stride]
// bf16 buffer (global or LDS).  ISA 16-bit A layout: lane<16 holds row m0+lane,
// K = {k0..k0+7, k0+16..k0+23}; lane>=16 holds row m0+lane-16,
// K = {k0+8..k0+15, k0+24..k0+31}.  B operands are loaded as the NxK transpose
// with the identical pattern (lane = output column).
static __device__ __forceinline__ v16bf load_frag(const unsigned short* base,
                                                  int stride, int m0, int k0,
                                                  int lh, int hi) {
  const unsigned short* p = base + (m0 + lh) * stride + k0 + hi * 8;
  BF16Frag f;
  f.u[0] = *reinterpret_cast<const u32x4*>(p);
  f.u[1] = *reinterpret_cast<const u32x4*>(p + 16);
  return f.v;
}

// Same fragment but sourced from an f32 LDS buffer (state matrix), converting.
static __device__ __forceinline__ v16bf load_frag_f32(const float* base,
                                                      int stride, int n0, int k0,
                                                      int lh, int hi) {
  const float* p = base + (n0 + lh) * stride + k0 + hi * 8;
  BF16Frag f;
#pragma unroll
  for (int t = 0; t < 8; ++t) {
    f.s[t]     = f2bf(p[t]);
    f.s[8 + t] = f2bf(p[16 + t]);
  }
  return f.v;
}

// CDNA5 async copy: one 16-byte chunk per lane, global -> LDS, ASYNCcnt-tracked.
static __device__ __forceinline__ void async_copy_b128(const unsigned short* g,
                                                       unsigned short* lds) {
  unsigned ldsoff = (unsigned)(uintptr_t)lds;
  asm volatile("global_load_async_to_lds_b128 %0, %1, off"
               :: "v"(ldsoff), "v"(g)
               : "memory");
}
static __device__ __forceinline__ void wait_async0() {
  asm volatile("s_wait_asynccnt 0x0" ::: "memory");
}

// Stage a 128x32 bf16 tile via per-lane async b128 copies (2 per thread).
static __device__ __forceinline__ void issue_tile_128x32(
    const unsigned short* __restrict__ src, int srcStride, int r0, int k0,
    unsigned short* dst, int tid) {
#pragma unroll
  for (int i = 0; i < 2; ++i) {
    int chunk = tid + i * 256;       // 0..511
    int r = chunk >> 2;              // row 0..127
    int c = (chunk & 3) * 8;         // element offset 0,8,16,24
    async_copy_b128(src + (size_t)(r0 + r) * srcStride + k0 + c,
                    dst + r * 32 + c);
  }
}

// Tensor Data Mover: DMA a 128x32 bf16 tile (rows x K-slab) from a row-major
// [rows][K] bf16 tensor into LDS.  D# built per CDNA5 ISA.
static __device__ __forceinline__ void tdm_load_tile_128x32(
    const unsigned short* gtile, unsigned short* ldsdst,
    unsigned K, unsigned rows) {
#ifdef HAVE_TDM
  unsigned long long ga = (unsigned long long)(uintptr_t)gtile;
  unsigned ldsoff = (unsigned)(uintptr_t)ldsdst;
  v4u g0;
  g0[0] = 1u;                                          // count=1, user D#
  g0[1] = ldsoff;                                      // lds_addr [63:32]
  g0[2] = (unsigned)(ga & 0xFFFFFFFFu);                // global_addr lo
  g0[3] = (unsigned)((ga >> 32) & 0x01FFFFFFu) | (2u << 30);  // addr hi, type=2
  v8i g1;
  g1[0] = (int)(1u << 16);                             // data_size=1 (2B)
  g1[1] = (int)((K & 0xFFFFu) << 16);                  // tensor_dim0 lo16
  g1[2] = (int)(((K >> 16) & 0xFFFFu) | ((rows & 0xFFFFu) << 16));  // dim0 hi | dim1 lo
  g1[3] = (int)(((rows >> 16) & 0xFFFFu) | (32u << 16));            // dim1 hi | tile_dim0=32
  g1[4] = 128;                                         // tile_dim1=128
  g1[5] = (int)K;                                      // tensor_dim0_stride lo
  g1[6] = 0;
  g1[7] = 0;
  v4i z = {0, 0, 0, 0};
#if __clang_major__ >= 23
  v8i z8 = {0, 0, 0, 0, 0, 0, 0, 0};
  __builtin_amdgcn_tensor_load_to_lds(g0, g1, z, z, z8, 0);
#else
  __builtin_amdgcn_tensor_load_to_lds(g0, g1, z, z, 0);
#endif
#else
  (void)gtile; (void)ldsdst; (void)K; (void)rows;
#endif
}
static __device__ __forceinline__ void wait_tensor0() {
#ifdef HAVE_TDM
  __builtin_amdgcn_s_wait_tensorcnt(0);
#endif
}

// ---------- pre-pass kernels ----------
__global__ void f32_to_bf16_kernel(const float* __restrict__ in,
                                   unsigned short* __restrict__ out, int n) {
  int i = blockIdx.x * blockDim.x + threadIdx.x;
  if (i < n) out[i] = f2bf(in[i]);
}

// W is [K][N] row-major fp32; produce Wt = bf16 [N][K].
__global__ void transpose_bf16_kernel(const float* __restrict__ W,
                                      unsigned short* __restrict__ Wt,
                                      int K, int N) {
  int idx = blockIdx.x * blockDim.x + threadIdx.x;
  if (idx >= K * N) return;
  int k = idx / N, n = idx % N;
  Wt[n * K + k] = f2bf(W[idx]);
}

// ---------- WMMA GEMM: async-LDS (A) + TDM (B) double buffering ----------
// out = act(A[MxK] * Bt[NxK]^T)
// mode 0: bf16 out (plain)  mode 1: bf16 out, relu(x)+1e-3  mode 2: f32 out + bias
// Block: 256 threads = 8 waves; block tile 128x128; wave tile 32x64.
__global__ __launch_bounds__(256) void gemm_bf16_kernel(
    const unsigned short* __restrict__ A, const unsigned short* __restrict__ Bt,
    void* __restrict__ out, const float* __restrict__ bias,
    int M, int N, int K, int mode) {
  __shared__ __align__(16) unsigned short sA[2][128 * 32];
  __shared__ __align__(16) unsigned short sB[2][128 * 32];

  const int tid = threadIdx.x;
  const int w = tid >> 5, lane = tid & 31, lh = lane & 15, hi = lane >> 4;
  const int wm = (w & 3) * 32;   // wave M offset within block tile (4 waves)
  const int wn = (w >> 2) * 64;  // wave N offset within block tile (2 waves)
  const int cb = blockIdx.x * 128;
  const int rb = blockIdx.y * 128;

  v8f acc[2][4];
#pragma unroll
  for (int mt = 0; mt < 2; ++mt)
#pragma unroll
    for (int nt = 0; nt < 4; ++nt)
      acc[mt][nt] = (v8f){0.f,0.f,0.f,0.f,0.f,0.f,0.f,0.f};

  const int KT = K / 32;
  issue_tile_128x32(A, K, rb, 0, sA[0], tid);      // A slab: per-lane async path
#ifdef HAVE_TDM
  if (w == 0)                                      // B slab: one TDM DMA per slab
    tdm_load_tile_128x32(Bt + (size_t)cb * K, sB[0], (unsigned)K, 128u);
#else
  issue_tile_128x32(Bt, K, cb, 0, sB[0], tid);
#endif

  int buf = 0;
  for (int kt = 0; kt < KT; ++kt) {
    wait_async0();
    if (w == 0) wait_tensor0();
    __syncthreads();                       // current buffers visible to all
    if (kt + 1 < KT) {                     // prefetch next slab (overlaps math)
      issue_tile_128x32(A, K, rb, (kt + 1) * 32, sA[buf ^ 1], tid);
#ifdef HAVE_TDM
      if (w == 0)
        tdm_load_tile_128x32(Bt + (size_t)cb * K + (kt + 1) * 32,
                             sB[buf ^ 1], (unsigned)K, 128u);
#else
      issue_tile_128x32(Bt, K, cb, (kt + 1) * 32, sB[buf ^ 1], tid);
#endif
    }
#pragma unroll
    for (int mt = 0; mt < 2; ++mt) {
      v16bf a = load_frag(sA[buf], 32, wm + mt * 16, 0, lh, hi);
#pragma unroll
      for (int nt = 0; nt < 4; ++nt) {
        v16bf b = load_frag(sB[buf], 32, wn + nt * 16, 0, lh, hi);
        acc[mt][nt] = __builtin_amdgcn_wmma_f32_16x16x32_bf16(
            false, a, false, b, (short)0, acc[mt][nt], false, false);
      }
    }
    __syncthreads();                       // done reading buf before reuse
    buf ^= 1;
  }

#pragma unroll
  for (int mt = 0; mt < 2; ++mt)
#pragma unroll
    for (int nt = 0; nt < 4; ++nt)
#pragma unroll
      for (int r = 0; r < 8; ++r) {
        int m = rb + wm + mt * 16 + r + hi * 8;
        int n = cb + wn + nt * 16 + lh;
        float v = acc[mt][nt][r];
        if (mode == 1) v = fmaxf(v, 0.f) + 1.0e-3f;
        if (mode == 2) {
          ((float*)out)[(size_t)m * N + n] = v + bias[n];
        } else {
          ((unsigned short*)out)[(size_t)m * N + n] = f2bf(v);
        }
      }
}

// ---------- chunked causal linear attention (one workgroup per head) ----------
__global__ __launch_bounds__(128) void linattn_kernel(
    const unsigned short* __restrict__ Qg, const unsigned short* __restrict__ Kg,
    const unsigned short* __restrict__ Vg, unsigned short* __restrict__ Og) {
  __shared__ __align__(16) unsigned short sQ[64 * 64];   // [token][d]
  __shared__ __align__(16) unsigned short sK[64 * 64];   // [token][d]
  __shared__ __align__(16) unsigned short sKT[64 * 64];  // [d][token]
  __shared__ __align__(16) unsigned short sVT[64 * 64];  // [e][token]
  __shared__ __align__(16) unsigned short sA[64 * 72];   // masked QK^T, padded
  __shared__ __align__(16) float sST[64 * 68];           // S^T: [e][d], padded
  __shared__ float sPrev[64];                            // running sum of k
  __shared__ float sD[64];                               // 1/denominator

  const int h = blockIdx.x;
  const int tid = threadIdx.x;
  const int w = tid >> 5, lane = tid & 31, lh = lane & 15, hi = lane >> 4;
  const int m0 = w * 16;

  for (int i = tid; i < 64 * 68; i += 128) sST[i] = 0.f;
  if (tid < 64) sPrev[tid] = 0.f;
  __syncthreads();

  for (int c = 0; c < NCH; ++c) {
    const int t0 = c * CH;
    for (int idx = tid; idx < 64 * 64; idx += 128) {
      int j = idx >> 6, d = idx & 63;
      size_t off = (size_t)(t0 + j) * DIMV + h * DH + d;
      unsigned short q = Qg[off], k = Kg[off], v = Vg[off];
      sQ[j * 64 + d]  = q;
      sK[j * 64 + d]  = k;
      sKT[d * 64 + j] = k;
      sVT[d * 64 + j] = v;
    }
    __syncthreads();

    v8f accA[4], accO[4];
#pragma unroll
    for (int nt = 0; nt < 4; ++nt) {
      accA[nt] = (v8f){0.f,0.f,0.f,0.f,0.f,0.f,0.f,0.f};
      accO[nt] = (v8f){0.f,0.f,0.f,0.f,0.f,0.f,0.f,0.f};
    }

    // A = Q K^T  and  O += Q * S_prev   (both K-dim = DH = 64)
    for (int k0 = 0; k0 < 64; k0 += 32) {
      v16bf a = load_frag(sQ, 64, m0, k0, lh, hi);
#pragma unroll
      for (int nt = 0; nt < 4; ++nt) {
        v16bf bK = load_frag(sK, 64, nt * 16, k0, lh, hi);
        accA[nt] = __builtin_amdgcn_wmma_f32_16x16x32_bf16(
            false, a, false, bK, (short)0, accA[nt], false, false);
        v16bf bS = load_frag_f32(sST, 68, nt * 16, k0, lh, hi);
        accO[nt] = __builtin_amdgcn_wmma_f32_16x16x32_bf16(
            false, a, false, bS, (short)0, accO[nt], false, false);
      }
    }
    // causal mask (inclusive: j <= i) and publish A to LDS
#pragma unroll
    for (int nt = 0; nt < 4; ++nt)
#pragma unroll
      for (int r = 0; r < 8; ++r) {
        int mi = m0 + r + hi * 8;
        int nj = nt * 16 + lh;
        sA[mi * 72 + nj] = f2bf((nj <= mi) ? accA[nt][r] : 0.f);
      }
    __syncthreads();

    // O += mask(A) * V   (K-dim = chunk tokens = 64)
    for (int k0 = 0; k0 < 64; k0 += 32) {
      v16bf a = load_frag(sA, 72, m0, k0, lh, hi);
#pragma unroll
      for (int nt = 0; nt < 4; ++nt) {
        v16bf bV = load_frag(sVT, 64, nt * 16, k0, lh, hi);
        accO[nt] = __builtin_amdgcn_wmma_f32_16x16x32_bf16(
            false, a, false, bV, (short)0, accO[nt], false, false);
      }
    }

    // denominator:  D_i = rowsum(mask(A))_i + q_i . s_prev
    if (tid < 64) {
      float s = 0.f;
      for (int j = 0; j < 64; ++j) s += bf2f(sA[tid * 72 + j]);
      for (int d = 0; d < 64; ++d) s += bf2f(sQ[tid * 64 + d]) * sPrev[d];
      sD[tid] = 1.0f / s;
    }
    __syncthreads();

    // S += K^T V  (store transposed into sST: next chunk's B-frags contiguous)
    {
      v8f accU[4];
#pragma unroll
      for (int nt = 0; nt < 4; ++nt)
        accU[nt] = (v8f){0.f,0.f,0.f,0.f,0.f,0.f,0.f,0.f};
      for (int k0 = 0; k0 < 64; k0 += 32) {
        v16bf a = load_frag(sKT, 64, m0, k0, lh, hi);
#pragma unroll
        for (int nt = 0; nt < 4; ++nt) {
          v16bf bV = load_frag(sVT, 64, nt * 16, k0, lh, hi);
          accU[nt] = __builtin_amdgcn_wmma_f32_16x16x32_bf16(
              false, a, false, bV, (short)0, accU[nt], false, false);
        }
      }
#pragma unroll
      for (int nt = 0; nt < 4; ++nt)
#pragma unroll
        for (int r = 0; r < 8; ++r) {
          int dd = m0 + r + hi * 8;        // row of S (feature d)
          int ee = nt * 16 + lh;           // col of S (feature e)
          sST[ee * 68 + dd] += accU[nt][r];
        }
    }
    if (tid < 64) {
      float s = 0.f;
      for (int j = 0; j < 64; ++j) s += bf2f(sKT[tid * 64 + j]);
      sPrev[tid] += s;
    }

    // scaled output
#pragma unroll
    for (int nt = 0; nt < 4; ++nt)
#pragma unroll
      for (int r = 0; r < 8; ++r) {
        int mi = m0 + r + hi * 8;
        int ee = nt * 16 + lh;
        float v = accO[nt][r] * sD[mi];
        Og[(size_t)(t0 + mi) * DIMV + h * DH + ee] = f2bf(v);
      }
    __syncthreads();
  }
}

// ---------- orchestration ----------
extern "C" void kernel_launch(void* const* d_in, const int* in_sizes, int n_in,
                              void* d_out, int out_size, void* d_ws, size_t ws_size,
                              hipStream_t stream) {
  (void)in_sizes; (void)n_in; (void)out_size; (void)ws_size;
  const float* x  = (const float*)d_in[0];
  const float* Wq = (const float*)d_in[1];
  const float* Wk = (const float*)d_in[2];
  const float* Wv = (const float*)d_in[3];
  const float* Wo = (const float*)d_in[4];
  const float* bo = (const float*)d_in[5];

  char* ws = (char*)d_ws;
  const size_t SZ_X = (size_t)NTOK * DIMV * 2;   // 4 MB bf16
  const size_t SZ_W = (size_t)DIMV * DIMV * 2;   // 2 MB bf16
  unsigned short* xb  = (unsigned short*)(ws);
  unsigned short* WqT = (unsigned short*)(ws + SZ_X);
  unsigned short* WkT = (unsigned short*)(ws + SZ_X + SZ_W);
  unsigned short* WvT = (unsigned short*)(ws + SZ_X + 2 * SZ_W);
  unsigned short* WoT = (unsigned short*)(ws + SZ_X + 3 * SZ_W);
  unsigned short* Qb  = (unsigned short*)(ws + SZ_X + 4 * SZ_W);
  unsigned short* Kb  = (unsigned short*)(ws + 2 * SZ_X + 4 * SZ_W);
  unsigned short* Vb  = (unsigned short*)(ws + 3 * SZ_X + 4 * SZ_W);
  unsigned short* Ab  = (unsigned short*)(ws + 4 * SZ_X + 4 * SZ_W);

  f32_to_bf16_kernel<<<(NTOK * DIMV + 255) / 256, 256, 0, stream>>>(x, xb, NTOK * DIMV);
  transpose_bf16_kernel<<<(DIMV * DIMV + 255) / 256, 256, 0, stream>>>(Wq, WqT, DIMV, DIMV);
  transpose_bf16_kernel<<<(DIMV * DIMV + 255) / 256, 256, 0, stream>>>(Wk, WkT, DIMV, DIMV);
  transpose_bf16_kernel<<<(DIMV * DIMV + 255) / 256, 256, 0, stream>>>(Wv, WvT, DIMV, DIMV);
  transpose_bf16_kernel<<<(DIMV * DIMV + 255) / 256, 256, 0, stream>>>(Wo, WoT, DIMV, DIMV);

  dim3 ggrid(DIMV / 128, NTOK / 128);  // (8, 16)
  gemm_bf16_kernel<<<ggrid, 256, 0, stream>>>(xb, WqT, Qb, nullptr, NTOK, DIMV, DIMV, 1);
  gemm_bf16_kernel<<<ggrid, 256, 0, stream>>>(xb, WkT, Kb, nullptr, NTOK, DIMV, DIMV, 1);
  gemm_bf16_kernel<<<ggrid, 256, 0, stream>>>(xb, WvT, Vb, nullptr, NTOK, DIMV, DIMV, 0);

  linattn_kernel<<<HEADS, 128, 0, stream>>>(Qb, Kb, Vb, Ab);

  gemm_bf16_kernel<<<ggrid, 256, 0, stream>>>(Ab, WoT, d_out, bo, NTOK, DIMV, DIMV, 2);
}
